// TokenFuse_48524540510953
// MI455X (gfx1250) — matrix operand order repaired and checked by reading, not compiled, not run on hardware
//
#include <hip/hip_runtime.h>
#include <hip/hip_bf16.h>
#include <math.h>

// ---------------- types for WMMA (CDNA5 / gfx1250, wave32) ----------------
typedef __bf16 v16bf __attribute__((ext_vector_type(16)));
typedef float  v8f   __attribute__((ext_vector_type(8)));

#define DIMC 128
#define HIDC 512
#define NTOK 32768              // 2*128*128 tokens
#define TM   64                 // tokens per workgroup in MLP kernel

#define FX 636.0779f
#define FY 635.2265f
#define CXc 634.8217f
#define CYc 357.0233f
#define SH 480
#define SW 640
#define NPIX (SH*SW)            // 307200
#define MAXD 10.0f

// Branch-free GELU (tanh form, native v_exp_f32). Saturates correctly at +/-inf.
__device__ __forceinline__ float tf_gelu(float x) {
    float u = 0.7978845608028654f * (x + 0.044715f * x * x * x);
    float e = __expf(2.0f * u);            // v_exp_f32, no EXEC branches
    float t = 1.0f - 2.0f / (e + 1.0f);    // tanh(u)
    return 0.5f * x * (1.0f + t);
}

// ---------------------------------------------------------------------------
// Kernel 1: weight prep (f32 -> bf16, conv weight transpose to [tap][C])
// ---------------------------------------------------------------------------
__global__ void tf_prep(const float* __restrict__ fc1w, const float* __restrict__ fc2w,
                        const float* __restrict__ convw,
                        __bf16* __restrict__ w1b, __bf16* __restrict__ w2b,
                        float* __restrict__ cwt) {
    int i = blockIdx.x * 256 + threadIdx.x;
    if (i < DIMC * HIDC) w1b[i] = (__bf16)fc1w[i];
    if (i < HIDC * DIMC) w2b[i] = (__bf16)fc2w[i];
    if (i < DIMC * 9) {
        int c = i / 9, k = i % 9;
        cwt[k * DIMC + c] = convw[i];   // [1,C,3,3] -> [9][C]
    }
}

// ---------------------------------------------------------------------------
// Kernel 2: fused MLP  (X*W1 + b1 -> GELU -> *W2 + b2)  via bf16 WMMA
//   grid: NTOK/TM workgroups, 256 threads (8 waves of 32)
//   dyn LDS: As[TM][128] bf16 (16KB) + Hs[TM][512] bf16 (64KB) = 80KB
// ---------------------------------------------------------------------------
__global__ void tf_mlp_wmma(const float* __restrict__ x,     // [NTOK,128]
                            const __bf16* __restrict__ w1,   // [128,512]
                            const float* __restrict__ b1,    // [512]
                            const __bf16* __restrict__ w2,   // [512,128]
                            const float* __restrict__ b2,    // [128]
                            float* __restrict__ feat)        // [NTOK,128]
{
    extern __shared__ char smem[];
    __bf16* As = (__bf16*)smem;                      // [TM][DIMC]
    __bf16* Hs = (__bf16*)(smem + TM * DIMC * 2);    // [TM][HIDC]

    const int tid  = threadIdx.x;
    const int lane = tid & 31;
    const int wave = tid >> 5;
    const int half = lane >> 4;          // 0: lanes 0-15, 1: lanes 16-31
    const int l16  = lane & 15;
    const int tok0 = blockIdx.x * TM;

    // --- stage A tile into LDS as bf16 (float4 loads) ---
    const float* xa = x + (size_t)tok0 * DIMC;
    for (int idx = tid * 4; idx < TM * DIMC; idx += 256 * 4) {
        float4 v = *(const float4*)(xa + idx);
        As[idx + 0] = (__bf16)v.x; As[idx + 1] = (__bf16)v.y;
        As[idx + 2] = (__bf16)v.z; As[idx + 3] = (__bf16)v.w;
    }
    __syncthreads();

    // --- GEMM1: [TM x 128] x [128 x 512] ; tiles 4(M) x 32(N) = 128 / 8 waves
    for (int t = wave; t < (TM/16) * (HIDC/16); t += 8) {
        const int mt = t / (HIDC/16), nt = t % (HIDC/16);
        const int mBase = mt * 16, nBase = nt * 16;
        float bias = b1[nBase + l16];
        v8f acc;
        #pragma unroll
        for (int r = 0; r < 8; ++r) acc[r] = bias;

        #pragma unroll
        for (int kt = 0; kt < DIMC / 32; ++kt) {
            const int kBase = kt * 32;
            // A frag (ISA 7.12.2): lane-half picks K 0-7 / 8-15, elems 8..15 -> K+16
            v16bf af;
            const __bf16* ar = As + (mBase + l16) * DIMC + kBase + half * 8;
            #pragma unroll
            for (int e = 0; e < 8; ++e)  af[e] = ar[e];
            #pragma unroll
            for (int e = 8; e < 16; ++e) af[e] = ar[8 + e];
            // B frag: column n = l16, elems in same K order as A
            v16bf bf;
            const __bf16* br = w1 + (size_t)(kBase + half * 8) * HIDC + nBase + l16;
            #pragma unroll
            for (int e = 0; e < 8; ++e)  bf[e] = br[(size_t)e * HIDC];
            #pragma unroll
            for (int e = 8; e < 16; ++e) bf[e] = br[(size_t)(8 + e) * HIDC];
            acc = __builtin_amdgcn_wmma_f32_16x16x32_bf16(
                      false, af, false, bf, (short)0, acc, false, false);
        }
        // branch-free GELU + store hidden to LDS (bf16)
        #pragma unroll
        for (int r = 0; r < 8; ++r)
            Hs[(mBase + r + 8 * half) * HIDC + nBase + l16] = (__bf16)tf_gelu(acc[r]);
    }
    __syncthreads();

    // --- GEMM2: [TM x 512] x [512 x 128] ; tiles 4(M) x 8(N) = 32 / 8 waves
    for (int t = wave; t < (TM/16) * (DIMC/16); t += 8) {
        const int mt = t / (DIMC/16), nt = t % (DIMC/16);
        const int mBase = mt * 16, nBase = nt * 16;
        float bias = b2[nBase + l16];
        v8f acc;
        #pragma unroll
        for (int r = 0; r < 8; ++r) acc[r] = bias;

        #pragma unroll 4
        for (int kt = 0; kt < HIDC / 32; ++kt) {
            const int kBase = kt * 32;
            v16bf af;
            const __bf16* ar = Hs + (mBase + l16) * HIDC + kBase + half * 8;
            #pragma unroll
            for (int e = 0; e < 8; ++e)  af[e] = ar[e];
            #pragma unroll
            for (int e = 8; e < 16; ++e) af[e] = ar[8 + e];
            v16bf bf;
            const __bf16* br = w2 + (size_t)(kBase + half * 8) * DIMC + nBase + l16;
            #pragma unroll
            for (int e = 0; e < 8; ++e)  bf[e] = br[(size_t)e * DIMC];
            #pragma unroll
            for (int e = 8; e < 16; ++e) bf[e] = br[(size_t)(8 + e) * DIMC];
            acc = __builtin_amdgcn_wmma_f32_16x16x32_bf16(
                      false, af, false, bf, (short)0, acc, false, false);
        }
        #pragma unroll
        for (int r = 0; r < 8; ++r)
            feat[(size_t)(tok0 + mBase + r + 8 * half) * DIMC + nBase + l16] = acc[r];
    }
}

// ---------------------------------------------------------------------------
// Kernel 3: 3x3x128 -> 1 conv (correlation) + sigmoid*10
// ---------------------------------------------------------------------------
__global__ void tf_conv_sig(const float* __restrict__ feat, const float* __restrict__ cwt,
                            float* __restrict__ dpred) {
    int p = blockIdx.x * blockDim.x + threadIdx.x;     // 2*128*128
    if (p >= 2 * 128 * 128) return;
    int xx0 = p & 127, yy0 = (p >> 7) & 127, b = p >> 14;
    const float* fb = feat + (size_t)b * 128 * 128 * DIMC;
    float s = 0.f;
    for (int dy = -1; dy <= 1; ++dy) {
        int yy = yy0 + dy; if (yy < 0 || yy >= 128) continue;
        for (int dx = -1; dx <= 1; ++dx) {
            int xx = xx0 + dx; if (xx < 0 || xx >= 128) continue;
            const float4* fv = (const float4*)(fb + ((size_t)yy * 128 + xx) * DIMC);
            const float4* wv = (const float4*)(cwt + ((dy + 1) * 3 + (dx + 1)) * DIMC);
            #pragma unroll 8
            for (int c4 = 0; c4 < DIMC / 4; ++c4) {
                float4 f = fv[c4], w = wv[c4];
                s += f.x * w.x + f.y * w.y + f.z * w.z + f.w * w.w;
            }
        }
    }
    dpred[p] = MAXD / (1.0f + __expf(-s));
}

// ---------------------------------------------------------------------------
// Kernel 4: nearest grid-sample of depth + integer index map
// ---------------------------------------------------------------------------
__global__ void tf_sample_depth(const float* __restrict__ coors, const float* __restrict__ dpred,
                                float* __restrict__ sdep, int* __restrict__ imap) {
    int p = blockIdx.x * 256 + threadIdx.x;            // 2*NPIX
    if (p >= 2 * NPIX) return;
    float gx = coors[2 * p], gy = coors[2 * p + 1];
    float ix = rintf(((gx + 1.f) * 128.f - 1.f) * 0.5f);   // round-to-even == jnp.round
    float iy = rintf(((gy + 1.f) * 128.f - 1.f) * 0.5f);
    bool valid = (ix >= 0.f) && (ix < 128.f) && (iy >= 0.f) && (iy < 128.f);
    int ixc = (int)fminf(fmaxf(ix, 0.f), 127.f);
    int iyc = (int)fminf(fmaxf(iy, 0.f), 127.f);
    int b = p / NPIX;
    int idx = iyc * 128 + ixc;
    sdep[p] = valid ? dpred[b * 16384 + idx] : 0.f;
    imap[p] = valid ? idx : -1;
}

// ---------------------------------------------------------------------------
// Kernel 5: SNE surface normals -> output channels 0..2
// ---------------------------------------------------------------------------
__device__ __forceinline__ void tf_fetch(const float* __restrict__ s, int i, int j,
                                         float& X, float& Y, float& Z, float& D) {
    float zr = 0.f;
    if (i >= 0 && i < SH && j >= 0 && j < SW) zr = s[i * SW + j];
    X = zr * ((float)j - CXc) * (1.f / FX);          // X from UNmasked Z (ref order)
    float Y0 = zr * ((float)i - CYc) * (1.f / FY);
    bool m = (Y0 <= 0.f);
    Z = m ? 0.f : zr;
    Y = m ? 0.f : Y0;
    D = (Z > 0.f) ? (1.f / Z) : 0.f;
}

__global__ void tf_sne(const float* __restrict__ sdep, float* __restrict__ out) {
    int p = blockIdx.x * 256 + threadIdx.x;            // 2*NPIX
    if (p >= 2 * NPIX) return;
    int b = p / NPIX, pix = p % NPIX;
    int i = pix / SW, j = pix % SW;
    const float* s = sdep + (size_t)b * NPIX;

    float X[3][3], Y[3][3], Z[3][3], D[3][3];
    #pragma unroll
    for (int da = -1; da <= 1; ++da)
        #pragma unroll
        for (int db = -1; db <= 1; ++db)
            tf_fetch(s, i + da, j + db, X[da+1][db+1], Y[da+1][db+1], Z[da+1][db+1], D[da+1][db+1]);

    float nx_t = (D[1][2] - D[1][0]) * FX;
    float ny_t = (D[2][1] - D[0][1]) * FY;
    float phi = atan2f(ny_t, nx_t);
    float ca = cosf(phi), sb = sinf(phi);
    float nxy2 = nx_t * nx_t + ny_t * ny_t;

    const int OFF[8][2] = {{-1,-1},{-1,0},{-1,1},{0,-1},{0,1},{1,-1},{1,0},{1,1}};
    float snx = 0.f, sny = 0.f, snz = 0.f;
    #pragma unroll
    for (int o = 0; o < 8; ++o) {
        int da = OFF[o][0] + 1, db = OFF[o][1] + 1;
        float Xd = X[1][1] - X[da][db];
        float Yd = Y[1][1] - Y[da][db];
        float Zd = Z[1][1] - Z[da][db];
        float nz = (Zd != 0.f) ? ((nx_t * Xd + ny_t * Yd) / Zd) : 0.f;
        float sq = nxy2 + nz * nz;
        if (Zd != 0.f && sq > 0.f) {
            float inv = rsqrtf(sq);
            snx += nx_t * inv; sny += ny_t * inv; snz += nz * inv;
        }
    }
    float num = snx * ca + sny * sb;
    float theta = (snz != 0.f) ? -atanf(num / snz) : 0.f;
    float st = sinf(theta);
    float nx = st * ca, ny = st * sb, nz = cosf(theta);
    float sg = (ny > 0.f) ? -1.f : 1.f;

    float* ob = out + (size_t)b * 131 * NPIX;
    ob[0 * NPIX + pix] = nx * sg;
    ob[1 * NPIX + pix] = ny * sg;
    ob[2 * NPIX + pix] = nz * sg;
}

// ---------------------------------------------------------------------------
// Kernel 6: seg-token nearest gather -> output channels 3..130
//   4 channels per thread: one float4 gather, 4 coalesced plane stores
// ---------------------------------------------------------------------------
__global__ void tf_seg_gather4(const float* __restrict__ seg, const int* __restrict__ imap,
                               float* __restrict__ out) {
    long long t = (long long)blockIdx.x * 256 + threadIdx.x;   // 2*(128/4)*NPIX
    if (t >= (long long)2 * (DIMC / 4) * NPIX) return;
    int j = (int)(t % SW); long long r = t / SW;
    int i = (int)(r % SH); r /= SH;
    int c4 = (int)(r % (DIMC / 4));
    int b  = (int)(r / (DIMC / 4));
    int idx = imap[b * NPIX + i * SW + j];
    float4 v = make_float4(0.f, 0.f, 0.f, 0.f);
    if (idx >= 0)
        v = *(const float4*)(seg + ((size_t)b * 16384 + idx) * DIMC + c4 * 4);
    size_t obase = ((size_t)b * 131 + 3 + c4 * 4) * NPIX + (size_t)i * SW + j;
    out[obase]            = v.x;
    out[obase + NPIX]     = v.y;
    out[obase + 2 * NPIX] = v.z;
    out[obase + 3 * NPIX] = v.w;
}

// ---------------------------------------------------------------------------
// launch
// ---------------------------------------------------------------------------
extern "C" void kernel_launch(void* const* d_in, const int* in_sizes, int n_in,
                              void* d_out, int out_size, void* d_ws, size_t ws_size,
                              hipStream_t stream) {
    const float* depth_token = (const float*)d_in[0];
    const float* seg_token   = (const float*)d_in[1];
    const float* coors       = (const float*)d_in[2];
    // d_in[3] = pos_emb (unused by the reference forward)
    const float* fc1w = (const float*)d_in[4];
    const float* fc1b = (const float*)d_in[5];
    const float* fc2w = (const float*)d_in[6];
    const float* fc2b = (const float*)d_in[7];
    const float* convw = (const float*)d_in[8];
    float* out = (float*)d_out;

    char* ws = (char*)d_ws;
    __bf16* w1b  = (__bf16*)(ws + 0);              // 128*512*2  = 131072
    __bf16* w2b  = (__bf16*)(ws + 131072);         // 512*128*2  = 131072
    float*  cwt  = (float*)(ws + 262144);          // 9*128*4    = 4608 (pad 8192)
    float*  feat = (float*)(ws + 270336);          // 32768*128*4 = 16777216
    float*  dpred= (float*)(ws + 17047552);        // 32768*4    = 131072
    float*  sdep = (float*)(ws + 17178624);        // 2*307200*4 = 2457600
    int*    imap = (int*)  (ws + 19636224);        // 2*307200*4 = 2457600

    tf_prep<<<256, 256, 0, stream>>>(fc1w, fc2w, convw, w1b, w2b, cwt);

    tf_mlp_wmma<<<NTOK / TM, 256, (TM * DIMC + TM * HIDC) * 2, stream>>>(
        depth_token, w1b, fc1b, w2b, fc2b, feat);

    tf_conv_sig<<<(2 * 128 * 128) / 256, 256, 0, stream>>>(feat, cwt, dpred);

    tf_sample_depth<<<(2 * NPIX) / 256, 256, 0, stream>>>(coors, dpred, sdep, imap);

    tf_sne<<<(2 * NPIX) / 256, 256, 0, stream>>>(sdep, out);

    tf_seg_gather4<<<(2 * (DIMC / 4) * NPIX) / 256, 256, 0, stream>>>(seg_token, imap, out);
}
